// AdaptiveSpatialAttention_83288005804462
// MI455X (gfx1250) — compile-verified, matrix operand order checked by reading, not run on hardware
//
#include <hip/hip_runtime.h>
#include <cstddef>

// ---------------------------------------------------------------------------
// AdaptiveSpatialAttention for MI455X (gfx1250, wave32, WMMA + TDM)
// B=4, H=W=160, L=25600, DIM=180, HEADS=6 (3 per branch), SPLIT=(8,16)
// ---------------------------------------------------------------------------

#define HRES 160
#define WRES 160
#define LTOK (HRES * WRES)   // 25600
#define BATCH 4
#define DIMC 180
#define HALFC 90
#define BRH 3                // heads per branch
#define HD 30                // head dim
#define NW_IMG 200           // windows per image
#define NPOS 465             // (2*8-1)*(2*16-1)
#define EPSF 1e-5f

typedef __attribute__((ext_vector_type(16))) _Float16 v16h;
typedef __attribute__((ext_vector_type(8)))  float    v8f;
typedef __attribute__((ext_vector_type(4)))  unsigned int v4u;
typedef __attribute__((ext_vector_type(8)))  int      v8i;
typedef __attribute__((ext_vector_type(4)))  int      v4i;

// ---------------------------------------------------------------------------
// WMMA helper (v_wmma_f32_16x16x32_f16)
// ---------------------------------------------------------------------------
__device__ __forceinline__ v8f wmma_f16(v16h a, v16h b, v8f c) {
  return __builtin_amdgcn_wmma_f32_16x16x32_f16(
      false, a, false, b, (short)0, c, false, false);
}

// Fragment index helper: K position for packed-pair v of this lane half.
__device__ __forceinline__ int frag_k(int v, int lane) {
  return ((v < 4) ? (2 * v) : (16 + 2 * (v - 4))) + (((lane >> 4) & 1) << 3);
}

// A fragment from f32 LDS tile (M-major, ld elems), converted + scaled.
__device__ __forceinline__ v16h frag_a_f32(const float* tile, int ld, int lane,
                                           float scale) {
  int m = lane & 15;
  v16h a;
#pragma unroll
  for (int v = 0; v < 8; ++v) {
    int k = frag_k(v, lane);
    a[2 * v]     = (_Float16)(tile[m * ld + k] * scale);
    a[2 * v + 1] = (_Float16)(tile[m * ld + k + 1] * scale);
  }
  return a;
}

// A fragment from f16 LDS tile (M-major).
__device__ __forceinline__ v16h frag_a_f16(const _Float16* tile, int ld,
                                           int lane) {
  int m = lane & 15;
  v16h a;
#pragma unroll
  for (int v = 0; v < 8; ++v) {
    int k = frag_k(v, lane);
    a[2 * v]     = tile[m * ld + k];
    a[2 * v + 1] = tile[m * ld + k + 1];
  }
  return a;
}

// B fragment from K-major f32 tile: (k, n) = tile[k*ld + n].
__device__ __forceinline__ v16h frag_b_f32(const float* tile, int ld, int lane) {
  int n = lane & 15;
  v16h b;
#pragma unroll
  for (int v = 0; v < 8; ++v) {
    int k = frag_k(v, lane);
    b[2 * v]     = (_Float16)tile[k * ld + n];
    b[2 * v + 1] = (_Float16)tile[(k + 1) * ld + n];
  }
  return b;
}

// B fragment from N-major f32 tile (transposed source): (k,n) = tile[n*ld+k].
__device__ __forceinline__ v16h frag_bt_f32(const float* tile, int ld, int lane) {
  int n = lane & 15;
  v16h b;
#pragma unroll
  for (int v = 0; v < 8; ++v) {
    int k = frag_k(v, lane);
    b[2 * v]     = (_Float16)tile[n * ld + k];
    b[2 * v + 1] = (_Float16)tile[n * ld + k + 1];
  }
  return b;
}

__device__ __forceinline__ float gelu_exact(float z) {
  return 0.5f * z * (1.f + erff(z * 0.70710678118654752f));
}
__device__ __forceinline__ float sigmoidf(float z) {
  return 1.f / (1.f + expf(-z));
}

// Low 32 bits of a generic pointer to LDS == LDS byte offset.
__device__ __forceinline__ unsigned lds_off(const void* p) {
  return (unsigned)(unsigned long long)(size_t)p;
}

// ---------------------------------------------------------------------------
// Tensor Data Mover: build D# (ISA 8.3-8.6) and issue TENSOR_LOAD_TO_LDS.
// 2D/3D tile of f32: tile dims (t0, t1, t2), tensor dims (d0, d1, d2) for
// OOB zero-fill, strides in elements (s0 between dim1 rows, s1 between dim2).
// t2 == 0 -> 2D.  6-arg builtin form (clang-23 / therock-10.0 lane).
// ---------------------------------------------------------------------------
__device__ __forceinline__ void tdm_load_f32(
    unsigned lds_byte, const float* g, unsigned t0, unsigned t1, unsigned t2,
    unsigned d0, unsigned d1, unsigned d2,
    unsigned long long s0, unsigned long long s1) {
  unsigned long long ga = (unsigned long long)(size_t)g;
  v4u g0;
  g0[0] = 1u;                                        // count=1 (valid, user)
  g0[1] = lds_byte;                                  // lds_addr
  g0[2] = (unsigned)(ga & 0xffffffffu);              // global_addr lo
  g0[3] = (unsigned)((ga >> 32) & 0x1ffffffu) | (2u << 30);  // hi | type=2
  v8i g1;
  g1[0] = (int)(2u << 16);                           // data_size=4B
  g1[1] = (int)((d0 & 0xffffu) << 16);               // tensor_dim0 lo16
  g1[2] = (int)(((d0 >> 16) & 0xffffu) | ((d1 & 0xffffu) << 16));
  g1[3] = (int)(((d1 >> 16) & 0xffffu) | ((t0 & 0xffffu) << 16)); // tile_dim0
  g1[4] = (int)((t1 & 0xffffu) | ((t2 & 0xffffu) << 16));  // tile_dim1/2
  g1[5] = (int)(unsigned)(s0 & 0xffffffffu);         // dim0_stride lo32
  g1[6] = (int)(((unsigned)(s0 >> 32) & 0xffffu) |
                (((unsigned)(s1 & 0xffffu)) << 16)); // dim0 hi | dim1 lo
  g1[7] = (int)(unsigned)((s1 >> 16) & 0xffffffffu); // dim1_stride hi32
  v4i g2;
  g2[0] = (int)d2;  // tensor_dim2
  g2[1] = 0;        // tensor_dim3 / lds_incr
  g2[2] = 0;        // tensor_dim2_stride lo (unused for 3D)
  g2[3] = 0;        // tensor_dim2_stride hi | tile_dim3
  v4i g3 = {0, 0, 0, 0};
  v8i g4 = {0, 0, 0, 0, 0, 0, 0, 0};
  __builtin_amdgcn_tensor_load_to_lds(g0, g1, g2, g3, g4, 0);
}

// ---------------------------------------------------------------------------
// Kernel 1: DynamicPosBias MLP (2 -> 5 -> 5 -> 5 -> 3 with LN+ReLU)
// ---------------------------------------------------------------------------
struct DynBranch {
  const float* proj_w; const float* proj_b;
  const float* ln_g[3]; const float* ln_b[3];
  const float* w[3];    const float* b[3];
};
struct DynParams { DynBranch br[2]; };

__global__ void dynpos_kernel(DynParams P, float* __restrict__ bias_tab) {
  int t = blockIdx.x * blockDim.x + threadIdx.x;
  if (t >= 2 * NPOS) return;
  int br = t / NPOS, p = t % NPOS;
  int hs = br ? 16 : 8, ws = br ? 8 : 16;
  int nbw = 2 * ws - 1;
  float bh = (float)((1 - hs) + p / nbw);
  float bw = (float)((1 - ws) + p % nbw);
  const DynBranch& q = P.br[br];
  float x[5];
#pragma unroll
  for (int d = 0; d < 5; ++d)
    x[d] = q.proj_b[d] + bh * q.proj_w[d * 2 + 0] + bw * q.proj_w[d * 2 + 1];
  for (int l = 0; l < 3; ++l) {
    float m = 0.f;
#pragma unroll
    for (int d = 0; d < 5; ++d) m += x[d];
    m *= 0.2f;
    float var = 0.f;
#pragma unroll
    for (int d = 0; d < 5; ++d) { float c = x[d] - m; var += c * c; }
    var *= 0.2f;
    float r = rsqrtf(var + EPSF);
    float h[5];
#pragma unroll
    for (int d = 0; d < 5; ++d) {
      float z = (x[d] - m) * r * q.ln_g[l][d] + q.ln_b[l][d];
      h[d] = fmaxf(z, 0.f);
    }
    int od = (l == 2) ? 3 : 5;
    float y[5];
    for (int o = 0; o < od; ++o) {
      float s = q.b[l][o];
#pragma unroll
      for (int d = 0; d < 5; ++d) s += h[d] * q.w[l][o * 5 + d];
      y[o] = s;
    }
    for (int o = 0; o < od; ++o) x[o] = y[o];
  }
#pragma unroll
  for (int h = 0; h < 3; ++h)
    bias_tab[br * NPOS * 3 + p * 3 + h] = x[h];
}

// ---------------------------------------------------------------------------
// Kernel 2: qkv = x @ qkv_w^T  (M=102400, K=180, N=540) — TDM + WMMA
// A tile (128x32) and B tile (64x32, N-major) DMA'd by TDM; the D#'s
// tensor_dim0=K-k0 / tensor_dim1=N-bn give free zero-fill of K/N tails.
// ---------------------------------------------------------------------------
__global__ __launch_bounds__(256) void qkv_gemm_kernel(
    const float* __restrict__ X, const float* __restrict__ Wq,
    float* __restrict__ Y) {
  __shared__ float Asf[128 * 32];   // packed by TDM, M-major
  __shared__ float Bsf[64 * 32];    // packed by TDM, N-major
  const int K = DIMC, N = 3 * DIMC;
  int bm = blockIdx.x * 128, bn = blockIdx.y * 64;
  int tid = threadIdx.x, wave = tid >> 5, lane = tid & 31;
  int wm = (wave >> 1) * 32, wn = (wave & 1) * 32;
  v8f acc[2][2] = {};
  for (int k0 = 0; k0 < K; k0 += 32) {
    if (wave == 0) {
      tdm_load_f32(lds_off(Asf), X + (size_t)bm * K + k0,
                   32, 128, 0, (unsigned)(K - k0), 128, 0, K, 0);
      __builtin_amdgcn_s_wait_tensorcnt(0);
    } else if (wave == 1) {
      tdm_load_f32(lds_off(Bsf), Wq + (size_t)bn * K + k0,
                   32, 64, 0, (unsigned)(K - k0), (unsigned)(N - bn), 0, K, 0);
      __builtin_amdgcn_s_wait_tensorcnt(0);
    }
    __syncthreads();
#pragma unroll
    for (int i = 0; i < 2; ++i) {
      v16h a = frag_a_f32(&Asf[(wm + 16 * i) * 32], 32, lane, 1.f);
#pragma unroll
      for (int j = 0; j < 2; ++j) {
        v16h b = frag_bt_f32(&Bsf[(wn + 16 * j) * 32], 32, lane);
        acc[i][j] = wmma_f16(a, b, acc[i][j]);
      }
    }
    __syncthreads();
  }
  int half = lane >> 4, c0 = lane & 15;
#pragma unroll
  for (int i = 0; i < 2; ++i)
#pragma unroll
    for (int j = 0; j < 2; ++j)
#pragma unroll
      for (int r = 0; r < 8; ++r) {
        int row = bm + wm + 16 * i + r + 8 * half;
        int col = bn + wn + 16 * j + c0;
        if (col < N) Y[(size_t)row * N + col] = acc[i][j][r];
      }
}

// ---------------------------------------------------------------------------
// Kernel 3: rectangular window attention; one block per (branch,b,window,head)
// Q/K/V tiles fetched by 3D TDM: tile (32, ws, hs), tensor dim0=30 -> the
// d=30..31 pad lanes are zero-filled by the data mover.
// ---------------------------------------------------------------------------
__global__ __launch_bounds__(256) void win_attn_kernel(
    const float* __restrict__ qkv, const float* __restrict__ bias_tab,
    float* __restrict__ attened) {
  extern __shared__ char smem[];
  float*    Qf = (float*)smem;                 // [128][32] f32
  float*    Kf = Qf + 128 * 32;                // [128][32]
  float*    Vf = Kf + 128 * 32;                // [128][32]
  float*    Ss = Vf + 128 * 32;                // [128][132] scores f32
  _Float16* Ps = (_Float16*)(Ss + 128 * 132);  // [128][136] probs f16

  int bid  = blockIdx.x;
  int br   = bid / (BATCH * NW_IMG * BRH);
  int rem  = bid % (BATCH * NW_IMG * BRH);
  int head = rem % BRH;
  int widx = rem / BRH;
  int b    = widx / NW_IMG;
  int wi   = widx % NW_IMG;
  int hs = br ? 16 : 8, ws = br ? 8 : 16;
  int nwx = WRES / ws;
  int wy = wi / nwx, wx = wi % nwx;
  int qoff = br * HALFC + head * HD;
  float scale = rsqrtf((float)HD);
  int tid = threadIdx.x, wave = tid >> 5, lane = tid & 31;
  int half = lane >> 4, nc = lane & 15;

  // Async 3D tile fetch of Q, K, V (one TDM descriptor per wave 0/1/2).
  {
    const float* base =
        qkv + ((size_t)b * LTOK + (size_t)(wy * hs) * WRES + wx * ws) *
                  (size_t)(3 * DIMC) + qoff;
    if (wave == 0) {
      tdm_load_f32(lds_off(Qf), base, 32, (unsigned)ws, (unsigned)hs,
                   HD, (unsigned)ws, (unsigned)hs,
                   3 * DIMC, (unsigned long long)(3 * DIMC) * WRES);
      __builtin_amdgcn_s_wait_tensorcnt(0);
    } else if (wave == 1) {
      tdm_load_f32(lds_off(Kf), base + DIMC, 32, (unsigned)ws, (unsigned)hs,
                   HD, (unsigned)ws, (unsigned)hs,
                   3 * DIMC, (unsigned long long)(3 * DIMC) * WRES);
      __builtin_amdgcn_s_wait_tensorcnt(0);
    } else if (wave == 2) {
      tdm_load_f32(lds_off(Vf), base + 2 * DIMC, 32, (unsigned)ws, (unsigned)hs,
                   HD, (unsigned)ws, (unsigned)hs,
                   3 * DIMC, (unsigned long long)(3 * DIMC) * WRES);
      __builtin_amdgcn_s_wait_tensorcnt(0);
    }
  }
  __syncthreads();

  // S = (Q*scale) @ K^T + bias ; wave -> 16-row strip, 8 column tiles.
  {
    int m0 = wave * 16;
    v16h a = frag_a_f32(&Qf[m0 * 32], 32, lane, scale);
#pragma unroll
    for (int nt = 0; nt < 8; ++nt) {
      v16h bf = frag_bt_f32(&Kf[(nt * 16) * 32], 32, lane);
      v8f acc = {};
      acc = wmma_f16(a, bf, acc);
#pragma unroll
      for (int r = 0; r < 8; ++r) {
        int i = m0 + r + 8 * half;
        int j = nt * 16 + nc;
        int tyi = i / ws, txi = i % ws;
        int tyj = j / ws, txj = j % ws;
        int rel = (tyi - tyj + hs - 1) * (2 * ws - 1) + (txi - txj + ws - 1);
        Ss[i * 132 + j] = acc[r] + bias_tab[br * NPOS * 3 + rel * 3 + head];
      }
    }
  }
  __syncthreads();

  // Row softmax -> f16 probabilities.
  if (tid < 128) {
    float mx = -3.4e38f;
    for (int j = 0; j < 128; ++j) mx = fmaxf(mx, Ss[tid * 132 + j]);
    float sum = 0.f;
    for (int j = 0; j < 128; ++j) sum += expf(Ss[tid * 132 + j] - mx);
    float inv = 1.f / sum;
    for (int j = 0; j < 128; ++j)
      Ps[tid * 136 + j] = (_Float16)(expf(Ss[tid * 132 + j] - mx) * inv);
  }
  __syncthreads();

  // O = P @ V ; wave -> 16-row strip x 2 N-tiles, 4 k-steps of 32.
  {
    int m0 = wave * 16;
    v8f acc[2] = {};
#pragma unroll
    for (int k0 = 0; k0 < 128; k0 += 32) {
      v16h a = frag_a_f16(&Ps[m0 * 136 + k0], 136, lane);
#pragma unroll
      for (int nt = 0; nt < 2; ++nt) {
        v16h bf = frag_b_f32(&Vf[k0 * 32 + nt * 16], 32, lane);
        acc[nt] = wmma_f16(a, bf, acc[nt]);
      }
    }
#pragma unroll
    for (int nt = 0; nt < 2; ++nt)
#pragma unroll
      for (int r = 0; r < 8; ++r) {
        int i = m0 + r + 8 * half;
        int d = nt * 16 + nc;
        if (d < HD) {
          int gy = wy * hs + i / ws, gx = wx * ws + i % ws;
          attened[((size_t)b * LTOK + gy * WRES + gx) * DIMC + qoff + d] =
              acc[nt][r];
        }
      }
  }
}

// ---------------------------------------------------------------------------
// Kernel 4: depthwise 3x3 conv on v + BN + GELU -> conv (B,C,H,W)
// ---------------------------------------------------------------------------
__global__ void dwconv_kernel(const float* __restrict__ qkv,
                              const float* __restrict__ dww,
                              const float* __restrict__ dwb,
                              const float* __restrict__ g,
                              const float* __restrict__ bb,
                              const float* __restrict__ mm,
                              const float* __restrict__ vv,
                              float* __restrict__ conv) {
  size_t idx = (size_t)blockIdx.x * 256 + threadIdx.x;
  if (idx >= (size_t)BATCH * DIMC * LTOK) return;
  int x = (int)(idx % WRES);
  size_t t = idx / WRES;
  int y = (int)(t % HRES); t /= HRES;
  int c = (int)(t % DIMC);
  int b = (int)(t / DIMC);
  float s = dwb[c];
#pragma unroll
  for (int dy = 0; dy < 3; ++dy)
#pragma unroll
    for (int dx = 0; dx < 3; ++dx) {
      int yy = y + dy - 1, xx = x + dx - 1;
      if (yy >= 0 && yy < HRES && xx >= 0 && xx < WRES)
        s += qkv[((size_t)b * LTOK + yy * WRES + xx) * (3 * DIMC) +
                 2 * DIMC + c] * dww[c * 9 + dy * 3 + dx];
    }
  float z = (s - mm[c]) * rsqrtf(vv[c] + EPSF) * g[c] + bb[c];
  conv[idx] = gelu_exact(z);
}

// ---------------------------------------------------------------------------
// Kernel 5a: global average pool of conv -> cm_in (B*C)
// ---------------------------------------------------------------------------
__global__ __launch_bounds__(256) void gap_kernel(const float* __restrict__ conv,
                                                  float* __restrict__ cm_in) {
  int bc = blockIdx.x;
  const float* p = conv + (size_t)bc * LTOK;
  float s = 0.f;
  for (int i = threadIdx.x; i < LTOK; i += 256) s += p[i];
  __shared__ float red[256];
  red[threadIdx.x] = s;
  __syncthreads();
  for (int k = 128; k > 0; k >>= 1) {
    if (threadIdx.x < k) red[threadIdx.x] += red[threadIdx.x + k];
    __syncthreads();
  }
  if (threadIdx.x == 0) cm_in[bc] = red[0] * (1.f / (float)LTOK);
}

// ---------------------------------------------------------------------------
// Kernel 5b: channel interaction MLP -> sigmoid gate (B*C)
// ---------------------------------------------------------------------------
__global__ __launch_bounds__(256) void chan_mlp_kernel(
    const float* __restrict__ cm_in, const float* __restrict__ w1,
    const float* __restrict__ b1, const float* __restrict__ g,
    const float* __restrict__ bb, const float* __restrict__ mm,
    const float* __restrict__ vv, const float* __restrict__ w2,
    const float* __restrict__ b2, float* __restrict__ gate) {
  __shared__ float h[BATCH][24];
  int tid = threadIdx.x;
  const int CH = DIMC / 8;  // 22
  if (tid < BATCH * CH) {
    int b = tid / CH, o = tid % CH;
    float s = b1[o];
    for (int c = 0; c < DIMC; ++c) s += cm_in[b * DIMC + c] * w1[o * DIMC + c];
    float z = (s - mm[o]) * rsqrtf(vv[o] + EPSF) * g[o] + bb[o];
    h[b][o] = gelu_exact(z);
  }
  __syncthreads();
  for (int i = tid; i < BATCH * DIMC; i += 256) {
    int b = i / DIMC, c = i % DIMC;
    float s = b2[c];
    for (int o = 0; o < CH; ++o) s += h[b][o] * w2[c * CH + o];
    gate[i] = sigmoidf(s);
  }
}

// ---------------------------------------------------------------------------
// Kernel 6: spatial interaction (1x1 -> BN -> GELU -> 1x1 -> sigmoid)
// ---------------------------------------------------------------------------
__global__ void spatial_kernel(const float* __restrict__ attened,
                               const float* __restrict__ w1,
                               const float* __restrict__ b1,
                               const float* __restrict__ g,
                               const float* __restrict__ bb,
                               const float* __restrict__ mm,
                               const float* __restrict__ vv,
                               const float* __restrict__ w2,
                               const float* __restrict__ b2,
                               float* __restrict__ gate) {
  size_t i = (size_t)blockIdx.x * 256 + threadIdx.x;
  if (i >= (size_t)BATCH * LTOK) return;
  const int CH = DIMC / 16;  // 11
  const float* row = attened + i * DIMC;
  float s[CH];
#pragma unroll
  for (int o = 0; o < CH; ++o) s[o] = b1[o];
  for (int c = 0; c < DIMC; c += 4) {
    float4 a = *(const float4*)&row[c];
#pragma unroll
    for (int o = 0; o < CH; ++o) {
      s[o] += a.x * w1[o * DIMC + c] + a.y * w1[o * DIMC + c + 1] +
              a.z * w1[o * DIMC + c + 2] + a.w * w1[o * DIMC + c + 3];
    }
  }
  float acc = b2[0];
#pragma unroll
  for (int o = 0; o < CH; ++o) {
    float z = (s[o] - mm[o]) * rsqrtf(vv[o] + EPSF) * g[o] + bb[o];
    acc += gelu_exact(z) * w2[o];
  }
  gate[i] = sigmoidf(acc);
}

// ---------------------------------------------------------------------------
// Kernel 7: out = (attened*sig(cm) + sig(sm)*conv) @ proj_w^T + proj_b
// A tile computed with float4 vector loads (K=180 is 16B aligned per row);
// B tile DMA'd by TDM with automatic K/N-tail zero-fill.
// ---------------------------------------------------------------------------
__global__ __launch_bounds__(256) void proj_gemm_kernel(
    const float* __restrict__ attened, const float* __restrict__ conv,
    const float* __restrict__ cgate, const float* __restrict__ sgate,
    const float* __restrict__ Wp, const float* __restrict__ pb,
    float* __restrict__ Y) {
  __shared__ _Float16 As[128][40];
  __shared__ float    Bsf[64 * 32];  // N-major, packed by TDM
  const int K = DIMC, N = DIMC;
  int bm = blockIdx.x * 128, bn = blockIdx.y * 64;
  int tid = threadIdx.x, wave = tid >> 5, lane = tid & 31;
  int wm = (wave >> 1) * 32, wn = (wave & 1) * 32;
  v8f acc[2][2] = {};
  for (int k0 = 0; k0 < K; k0 += 32) {
    if (wave == 1) {
      tdm_load_f32(lds_off(Bsf), Wp + (size_t)bn * K + k0,
                   32, 64, 0, (unsigned)(K - k0), (unsigned)(N - bn), 0, K, 0);
      __builtin_amdgcn_s_wait_tensorcnt(0);
    }
    if (k0 + 32 < K)  // global_prefetch_b8 of the next K slice of attened
      __builtin_prefetch(&attened[(size_t)(bm + (tid >> 1)) * K + k0 + 32],
                         0, 1);
    // Gated A tile: 1024 float4 slots, 4 per thread. K%4==0 so a slot is
    // either fully in range or fully out.
    for (int i = tid; i < 128 * 8; i += 256) {
      int r = i >> 3, c4 = (i & 7) * 4;
      int m = bm + r, k = k0 + c4;
      float4 av = make_float4(0.f, 0.f, 0.f, 0.f);
      if (k < K) {
        int b = m / LTOK, hw = m % LTOK;
        float4 at = *(const float4*)&attened[(size_t)m * DIMC + k];
        float4 cg = *(const float4*)&cgate[b * DIMC + k];
        float sg = sgate[(size_t)b * LTOK + hw];
        const float* cvp = &conv[((size_t)b * DIMC + k) * LTOK + hw];
        av.x = at.x * cg.x + sg * cvp[0];
        av.y = at.y * cg.y + sg * cvp[LTOK];
        av.z = at.z * cg.z + sg * cvp[2 * LTOK];
        av.w = at.w * cg.w + sg * cvp[3 * LTOK];
      }
      As[r][c4 + 0] = (_Float16)av.x;
      As[r][c4 + 1] = (_Float16)av.y;
      As[r][c4 + 2] = (_Float16)av.z;
      As[r][c4 + 3] = (_Float16)av.w;
    }
    __syncthreads();
#pragma unroll
    for (int i = 0; i < 2; ++i) {
      v16h a = frag_a_f16(&As[wm + 16 * i][0], 40, lane);
#pragma unroll
      for (int j = 0; j < 2; ++j) {
        v16h b = frag_bt_f32(&Bsf[(wn + 16 * j) * 32], 32, lane);
        acc[i][j] = wmma_f16(a, b, acc[i][j]);
      }
    }
    __syncthreads();
  }
  int half = lane >> 4, c0 = lane & 15;
#pragma unroll
  for (int i = 0; i < 2; ++i)
#pragma unroll
    for (int j = 0; j < 2; ++j)
#pragma unroll
      for (int r = 0; r < 8; ++r) {
        int row = bm + wm + 16 * i + r + 8 * half;
        int col = bn + wn + 16 * j + c0;
        if (col < N) Y[(size_t)row * N + col] = acc[i][j][r] + pb[col];
      }
}

// ---------------------------------------------------------------------------
// Host-side orchestration
// ---------------------------------------------------------------------------
extern "C" void kernel_launch(void* const* d_in, const int* in_sizes, int n_in,
                              void* d_out, int out_size, void* d_ws,
                              size_t ws_size, hipStream_t stream) {
  (void)in_sizes; (void)n_in; (void)out_size; (void)ws_size;
  const float* x      = (const float*)d_in[0];
  const float* qkv_w  = (const float*)d_in[1];
  const float* proj_w = (const float*)d_in[2];
  const float* proj_b = (const float*)d_in[3];
  DynParams dp;
  for (int br = 0; br < 2; ++br) {
    int base = 4 + br * 14;
    dp.br[br].proj_w = (const float*)d_in[base + 0];
    dp.br[br].proj_b = (const float*)d_in[base + 1];
    for (int l = 0; l < 3; ++l) {
      dp.br[br].ln_g[l] = (const float*)d_in[base + 2 + l * 4];
      dp.br[br].ln_b[l] = (const float*)d_in[base + 3 + l * 4];
      dp.br[br].w[l]    = (const float*)d_in[base + 4 + l * 4];
      dp.br[br].b[l]    = (const float*)d_in[base + 5 + l * 4];
    }
  }
  const float* dw_w   = (const float*)d_in[32];
  const float* dw_b   = (const float*)d_in[33];
  const float* dwbn_g = (const float*)d_in[34];
  const float* dwbn_b = (const float*)d_in[35];
  const float* dwbn_m = (const float*)d_in[36];
  const float* dwbn_v = (const float*)d_in[37];
  const float* cibn_g = (const float*)d_in[38];
  const float* cibn_b = (const float*)d_in[39];
  const float* cibn_m = (const float*)d_in[40];
  const float* cibn_v = (const float*)d_in[41];
  const float* sibn_g = (const float*)d_in[42];
  const float* sibn_b = (const float*)d_in[43];
  const float* sibn_m = (const float*)d_in[44];
  const float* sibn_v = (const float*)d_in[45];
  const float* ci_w1  = (const float*)d_in[46];
  const float* ci_b1  = (const float*)d_in[47];
  const float* ci_w2  = (const float*)d_in[48];
  const float* ci_b2  = (const float*)d_in[49];
  const float* si_w1  = (const float*)d_in[50];
  const float* si_b1  = (const float*)d_in[51];
  const float* si_w2  = (const float*)d_in[52];
  const float* si_b2  = (const float*)d_in[53];

  float* ws = (float*)d_ws;
  size_t o = 0;
  float* qkv      = ws + o; o += (size_t)BATCH * LTOK * 3 * DIMC;  // 221 MB
  float* attened  = ws + o; o += (size_t)BATCH * LTOK * DIMC;      // 74 MB
  float* conv     = ws + o; o += (size_t)BATCH * LTOK * DIMC;      // 74 MB
  float* cm_in    = ws + o; o += BATCH * DIMC;
  float* cgate    = ws + o; o += BATCH * DIMC;
  float* sgate    = ws + o; o += (size_t)BATCH * LTOK;
  float* bias_tab = ws + o; o += 2 * NPOS * 3;

  dynpos_kernel<<<2, 512, 0, stream>>>(dp, bias_tab);

  qkv_gemm_kernel<<<dim3(800, 9), 256, 0, stream>>>(x, qkv_w, qkv);

  const size_t attn_lds =
      (size_t)3 * 128 * 32 * sizeof(float)     // Q,K,V f32 tiles
      + (size_t)128 * 132 * sizeof(float)      // scores
      + (size_t)128 * 136 * sizeof(_Float16);  // probs
  win_attn_kernel<<<2 * BATCH * NW_IMG * BRH, 256, attn_lds, stream>>>(
      qkv, bias_tab, attened);

  dwconv_kernel<<<(BATCH * DIMC * LTOK) / 256, 256, 0, stream>>>(
      qkv, dw_w, dw_b, dwbn_g, dwbn_b, dwbn_m, dwbn_v, conv);

  gap_kernel<<<BATCH * DIMC, 256, 0, stream>>>(conv, cm_in);
  chan_mlp_kernel<<<1, 256, 0, stream>>>(cm_in, ci_w1, ci_b1, cibn_g, cibn_b,
                                         cibn_m, cibn_v, ci_w2, ci_b2, cgate);

  spatial_kernel<<<(BATCH * LTOK) / 256, 256, 0, stream>>>(
      attened, si_w1, si_b1, sibn_g, sibn_b, sibn_m, sibn_v, si_w2, si_b2,
      sgate);

  proj_gemm_kernel<<<dim3(800, 3), 256, 0, stream>>>(
      attened, conv, cgate, sgate, proj_w, proj_b, (float*)d_out);
}
